// GineEncoder_44564580663333
// MI455X (gfx1250) — matrix operand order repaired and checked by reading, not compiled, not run on hardware
//
#include <hip/hip_runtime.h>
#include <hip/hip_bf16.h>
#include <stdint.h>

// ---------------- constants -----------------
#define DD   300          // feature dim
#define DP   320          // padded dim (multiple of 32 for WMMA K, 64 for N tiles)
#define KT   32           // WMMA K step (f16)
#define BM   256          // block tile M (8 waves x 32 rows)
#define BN   64           // block tile N (4 WMMA col tiles)
#define LDA  48           // LDS row stride (elements): 96B rows, 16B aligned
#define LDB  48
#define LVL  5            // GNN layers
#define BN_EPS 1e-5f
#define NKT  (DP / KT)    // 10 K-steps

typedef __attribute__((ext_vector_type(16))) _Float16 v16h;
typedef __attribute__((ext_vector_type(8)))  _Float16 v8h;
typedef __attribute__((ext_vector_type(8)))  float    v8f;

static __device__ __forceinline__ v16h cat16(v8h lo, v8h hi) {
  return __builtin_shufflevector(lo, hi, 0,1,2,3,4,5,6,7,8,9,10,11,12,13,14,15);
}

// ---------------- weight pack: W[300x300] row-major -> Wt_f16[320][320] = W^T, zero padded
__global__ __launch_bounds__(256)
void pack_wt_kernel(const float* __restrict__ W, _Float16* __restrict__ Wt) {
  int idx = blockIdx.x * 256 + threadIdx.x;
  if (idx >= DP * DP) return;
  int n = idx / DP, k = idx % DP;
  float v = (n < DD && k < DD) ? W[k * DD + n] : 0.f;
  Wt[(size_t)n * DP + k] = (_Float16)v;
}

// ---------------- node first-half MLP: hid = relu([chir,fc] @ nW1 + nb1) -> f16 [N][DP]
__global__ __launch_bounds__(256)
void node_hidden_kernel(const float* __restrict__ chir, const float* __restrict__ fc,
                        const float* __restrict__ nW1, const float* __restrict__ nb1,
                        _Float16* __restrict__ out, int N) {
  int idx = blockIdx.x * 256 + threadIdx.x;
  if (idx >= N * DP) return;
  int n = idx / DP, d = idx % DP;
  float v = 0.f;
  if (d < DD) {
    v = chir[n] * nW1[d] + fc[n] * nW1[DD + d] + nb1[d];
    v = fmaxf(v, 0.f);
  }
  out[idx] = (_Float16)v;
}

// ---------------- edge first-half MLP: hid = relu(edge_attr @ eW1 + eb1) -> f16 [E][DP]
__global__ __launch_bounds__(256)
void edge_hidden_kernel(const float* __restrict__ eattr, const float* __restrict__ eW1,
                        const float* __restrict__ eb1, _Float16* __restrict__ out, int E) {
  int idx = blockIdx.x * 256 + threadIdx.x;
  if (idx >= E * DP) return;
  int e = idx / DP, d = idx % DP;
  float v = 0.f;
  if (d < DD) {
    const float* a = eattr + (size_t)e * 3;
    v = a[0] * eW1[d] + a[1] * eW1[DD + d] + a[2] * eW1[2 * DD + d] + eb1[d];
    v = fmaxf(v, 0.f);
  }
  out[idx] = (_Float16)v;
}

// ---------------- generic WMMA GEMM: C[M,300] = act(A_f16[M,320] @ Wt^T + bias)
// Wt is [320 n][320 k] (= W transposed, zero padded). Double-buffered LDS,
// one barrier per K-step, 32x64 output tile per wave (2x4 WMMA tiles).
// FULL=true  -> interior blocks, no M-edge guards in the hot loop.
// FULL=false -> single trailing block row handles the M remainder.
template <bool FULL>
__global__ __launch_bounds__(256)
void gemm_f16_kernel(const _Float16* __restrict__ A, const _Float16* __restrict__ Wt,
                     const float* __restrict__ bias, float* __restrict__ outF,
                     _Float16* __restrict__ outH, int M, int relu_flag, int blockOff) {
  __shared__ _Float16 lA[2][BM * LDA];
  __shared__ _Float16 lB[2][BN * LDB];

  const int tid   = threadIdx.x;
  const int lane  = tid & 31;
  const int wave  = tid >> 5;          // 0..7 -> 32-row strip
  const int mBase = (blockIdx.x + blockOff) * BM;
  const int nBase = blockIdx.y * BN;
  const int rHalf = lane >> 4;         // 0/1
  const int rMod  = lane & 15;

  const int rA   = tid >> 2;           // 0..63
  const int kOff = (tid & 3) * 8;      // 0,8,16,24

  const v8h zero8 = {};
  v8f acc[2][4] = {};
  v8h regA[4];
  v8h regB;

  // ---- preload K-tile 0 straight to LDS ----
#pragma unroll
  for (int q = 0; q < 4; ++q) {
    int r = rA + q * 64;
    int grow = mBase + r;
    v8h v = zero8;
    if (FULL || grow < M) v = *(const v8h*)(A + (size_t)grow * DP + kOff);
    *(v8h*)&lA[0][r * LDA + kOff] = v;
  }
  *(v8h*)&lB[0][rA * LDB + kOff] =
      *(const v8h*)(Wt + (size_t)(nBase + rA) * DP + kOff);
  __syncthreads();

  for (int kt = 0; kt < NKT; ++kt) {
    const int cur = kt & 1, nxt = cur ^ 1;

    // issue next K-tile's global loads early (stay in flight under WMMA)
    if (kt + 1 < NKT) {
      const int k0 = (kt + 1) * KT;
#pragma unroll
      for (int q = 0; q < 4; ++q) {
        int grow = mBase + rA + q * 64;
        regA[q] = zero8;
        if (FULL || grow < M) regA[q] = *(const v8h*)(A + (size_t)grow * DP + k0 + kOff);
      }
      regB = *(const v8h*)(Wt + (size_t)(nBase + rA) * DP + k0 + kOff);
    }

    // assemble fragments per CDNA5 16-bit WMMA layouts and multiply
    v16h aF[2], bF[4];
#pragma unroll
    for (int i = 0; i < 2; ++i) {
      int row = wave * 32 + i * 16 + rMod;        // A: lane -> row, K split by lane half
      v8h lo = *(const v8h*)&lA[cur][row * LDA + rHalf * 8];
      v8h hi = *(const v8h*)&lA[cur][row * LDA + 16 + rHalf * 8];
      aF[i] = cat16(lo, hi);
    }
#pragma unroll
    for (int j = 0; j < 4; ++j) {
      int col = j * 16 + rMod;                    // B: lane -> column, K split by lane half
      v8h lo = *(const v8h*)&lB[cur][col * LDB + rHalf * 8];
      v8h hi = *(const v8h*)&lB[cur][col * LDB + 16 + rHalf * 8];
      bF[j] = cat16(lo, hi);
    }
#pragma unroll
    for (int i = 0; i < 2; ++i)
#pragma unroll
      for (int j = 0; j < 4; ++j)
        acc[i][j] = __builtin_amdgcn_wmma_f32_16x16x32_f16(
            false, aF[i], false, bF[j], (short)0, acc[i][j], false, false);

    // commit next K-tile into the alternate buffer; single barrier per step
    if (kt + 1 < NKT) {
#pragma unroll
      for (int q = 0; q < 4; ++q)
        *(v8h*)&lA[nxt][(rA + q * 64) * LDA + kOff] = regA[q];
      *(v8h*)&lB[nxt][rA * LDB + kOff] = regB;
    }
    __syncthreads();
  }

  // epilogue: C 16x16 f32 layout => lane col = lane&15, VGPR v row = v + 8*(lane>>4)
#pragma unroll
  for (int i = 0; i < 2; ++i) {
#pragma unroll
    for (int j = 0; j < 4; ++j) {
      int col = nBase + j * 16 + rMod;
      float bv = (bias != nullptr && col < DD) ? bias[col] : 0.f;
#pragma unroll
      for (int v = 0; v < 8; ++v) {
        int row = mBase + wave * 32 + i * 16 + rHalf * 8 + v;
        if (!FULL && row >= M) continue;
        float c = acc[i][j][v] + bv;
        if (relu_flag) c = fmaxf(c, 0.f);
        if (outF != nullptr && col < DD) outF[(size_t)row * DD + col] = c;
        if (outH != nullptr) outH[(size_t)row * DP + col] = (col < DD) ? (_Float16)c : (_Float16)0.f;
      }
    }
  }
}

// ---------------- h += atom_emb[z]
__global__ __launch_bounds__(256)
void add_emb_kernel(float* __restrict__ h, const int* __restrict__ z,
                    const float* __restrict__ emb, int N) {
  int idx = blockIdx.x * 256 + threadIdx.x;
  if (idx >= N * DD) return;
  int n = idx / DD, d = idx % DD;
  h[idx] += emb[(size_t)z[n] * DD + d];
}

// ---------------- copy fp32
__global__ __launch_bounds__(256)
void copy_f32_kernel(const float* __restrict__ src, float* __restrict__ dst, int total) {
  int idx = blockIdx.x * 256 + threadIdx.x;
  if (idx < total) dst[idx] = src[idx];
}

// ---------------- zero fp32
__global__ __launch_bounds__(256)
void zero_f32_kernel(float* __restrict__ p, int total) {
  int idx = blockIdx.x * 256 + threadIdx.x;
  if (idx < total) p[idx] = 0.f;
}

// ---------------- message scatter: agg[dst] += relu(h[src] + e), 4 cols/thread
__global__ __launch_bounds__(256)
void scatter_msg_kernel(const float* __restrict__ h, const float* __restrict__ e,
                        const int* __restrict__ src, const int* __restrict__ dst,
                        float* __restrict__ agg, int E) {
  int t = blockIdx.x * 256 + threadIdx.x;
  const int CH = DD / 4;               // 75 chunks of 4
  if (t >= E * CH) return;
  int eid = t / CH;
  int c0  = (t % CH) * 4;
  int s = src[eid], d = dst[eid];
  const float* hs = h + (size_t)s * DD + c0;
  const float* ep = e + (size_t)eid * DD + c0;
  float* ap = agg + (size_t)d * DD + c0;
#pragma unroll
  for (int i = 0; i < 4; ++i) {
    float m = fmaxf(hs[i] + ep[i], 0.f);
    atomicAdd(&ap[i], m);
  }
}

// ---------------- fp32 [N,300] -> f16 [N,320] (zero pad)
__global__ __launch_bounds__(256)
void to_f16_kernel(const float* __restrict__ x, _Float16* __restrict__ out, int N) {
  int idx = blockIdx.x * 256 + threadIdx.x;
  if (idx >= N * DP) return;
  int n = idx / DP, d = idx % DP;
  out[idx] = (d < DD) ? (_Float16)x[(size_t)n * DD + d] : (_Float16)0.f;
}

// ---------------- BN stats: st[0..299]=sum, st[300..599]=sumsq
__global__ __launch_bounds__(256)
void bn_stats_kernel(const float* __restrict__ h2, float* __restrict__ st, int N) {
  int rbase = blockIdx.x * 128;
  int c0 = threadIdx.x;        // < 256 < 300, always valid
  int c1 = threadIdx.x + 256;  // valid if < 300
  float s0 = 0.f, q0 = 0.f, s1 = 0.f, q1 = 0.f;
  int rend = N - rbase; if (rend > 128) rend = 128;
  for (int r = 0; r < rend; ++r) {
    const float* rowp = h2 + (size_t)(rbase + r) * DD;
    float v0 = rowp[c0];
    s0 += v0; q0 += v0 * v0;
    if (c1 < DD) { float v1 = rowp[c1]; s1 += v1; q1 += v1 * v1; }
  }
  atomicAdd(&st[c0], s0);
  atomicAdd(&st[DD + c0], q0);
  if (c1 < DD) { atomicAdd(&st[c1], s1); atomicAdd(&st[DD + c1], q1); }
}

// ---------------- BN apply + relu -> h
__global__ __launch_bounds__(256)
void bn_apply_kernel(const float* __restrict__ h2, const float* __restrict__ st,
                     const float* __restrict__ gamma, const float* __restrict__ beta,
                     float* __restrict__ h, int N, float invN) {
  int idx = blockIdx.x * 256 + threadIdx.x;
  if (idx >= N * DD) return;
  int c = idx % DD;
  float mu = st[c] * invN;
  float var = st[DD + c] * invN - mu * mu;
  float v = (h2[idx] - mu) * rsqrtf(var + BN_EPS) * gamma[c] + beta[c];
  h[idx] = fmaxf(v, 0.f);
}

// ---------------- per-graph count
__global__ __launch_bounds__(256)
void pool_count_kernel(const int* __restrict__ batch, float* __restrict__ gcnt, int N) {
  int t = blockIdx.x * 256 + threadIdx.x;
  if (t < N) atomicAdd(&gcnt[batch[t]], 1.f);
}

// ---------------- per-graph sum
__global__ __launch_bounds__(256)
void pool_sum_kernel(const float* __restrict__ h, const int* __restrict__ batch,
                     float* __restrict__ gsum, int N) {
  int t = blockIdx.x * 256 + threadIdx.x;
  const int CH = DD / 4;
  if (t >= N * CH) return;
  int n = t / CH;
  int c0 = (t % CH) * 4;
  int g = batch[n];
  const float* hp = h + (size_t)n * DD + c0;
  float* gp = gsum + (size_t)g * DD + c0;
#pragma unroll
  for (int i = 0; i < 4; ++i) atomicAdd(&gp[i], hp[i]);
}

// ---------------- pooled mean -> f16 [G,320]
__global__ __launch_bounds__(256)
void pool_div_kernel(const float* __restrict__ gsum, const float* __restrict__ gcnt,
                     _Float16* __restrict__ out, int G) {
  int idx = blockIdx.x * 256 + threadIdx.x;
  if (idx >= G * DP) return;
  int g = idx / DP, k = idx % DP;
  float v = 0.f;
  if (k < DD) v = gsum[(size_t)g * DD + k] / fmaxf(gcnt[g], 1.f);
  out[idx] = (_Float16)v;
}

// ======================= host launcher =======================
static inline int cdiv(int a, int b) { return (a + b - 1) / b; }

extern "C" void kernel_launch(void* const* d_in, const int* in_sizes, int n_in,
                              void* d_out, int out_size, void* d_ws, size_t ws_size,
                              hipStream_t stream) {
  const int N = in_sizes[0];
  const int E = in_sizes[3] / 2;
  const int G = out_size / DD;

  const int*   z     = (const int*)d_in[0];
  const float* chir  = (const float*)d_in[1];
  const float* fch   = (const float*)d_in[2];
  const int*   eidx  = (const int*)d_in[3];
  const float* eattr = (const float*)d_in[4];
  const int*   batch = (const int*)d_in[5];
  const float* aemb  = (const float*)d_in[7];
  const float* nW1   = (const float*)d_in[8];
  const float* nb1   = (const float*)d_in[9];
  const float* nW2   = (const float*)d_in[10];
  const float* nb2   = (const float*)d_in[11];
  const float* eW1   = (const float*)d_in[12];
  const float* eb1   = (const float*)d_in[13];
  const float* eW2   = (const float*)d_in[14];
  const float* eb2   = (const float*)d_in[15];
  const float* mW1   = (const float*)d_in[16];  // [5][300][300]
  const float* mb1   = (const float*)d_in[17];  // [5][300]
  const float* mW2   = (const float*)d_in[18];
  const float* mb2   = (const float*)d_in[19];
  const float* gamma = (const float*)d_in[20];
  const float* beta  = (const float*)d_in[21];
  const float* pW    = (const float*)d_in[22];
  const float* pb    = (const float*)d_in[23];
  float* out = (float*)d_out;

  // -------- workspace carve (256B aligned) --------
  char* p = (char*)d_ws;
  auto carve = [&](size_t bytes) -> char* {
    char* r = p;
    p += (bytes + 255) & ~(size_t)255;
    return r;
  };
  const size_t WSZ = (size_t)DP * DP;          // elements per packed weight
  _Float16* wt   = (_Float16*)carve(13 * WSZ * sizeof(_Float16));
  float*    h    = (float*)carve((size_t)N * DD * sizeof(float));
  float*    xin  = (float*)carve((size_t)N * DD * sizeof(float));   // agg input / h2 reuse
  float*    ef   = (float*)carve((size_t)E * DD * sizeof(float));   // edge features fp32
  int maxME = (N > E) ? N : E;
  _Float16* bf1  = (_Float16*)carve((size_t)maxME * DP * sizeof(_Float16));
  _Float16* bf2  = (_Float16*)carve((size_t)N * DP * sizeof(_Float16));
  float*    gsum = (float*)carve((size_t)G * DD * sizeof(float));
  float*    gcnt = (float*)carve((size_t)G * sizeof(float));
  float*    st   = (float*)carve(2 * DD * sizeof(float));
  _Float16* pbuf = (_Float16*)carve((size_t)G * DP * sizeof(_Float16));

  const int* src = eidx;
  const int* dst = eidx + E;

  // GEMM dispatch: guard-free interior blocks + one trailing block row for the remainder
  auto launch_gemm = [&](const _Float16* A, const _Float16* W, const float* bias,
                         float* oF, _Float16* oH, int M, int relu) {
    int fullBlocks = M / BM;
    if (fullBlocks > 0)
      gemm_f16_kernel<true><<<dim3(fullBlocks, DP / BN), 256, 0, stream>>>(
          A, W, bias, oF, oH, M, relu, 0);
    if (M % BM)
      gemm_f16_kernel<false><<<dim3(1, DP / BN), 256, 0, stream>>>(
          A, W, bias, oF, oH, M, relu, fullBlocks);
  };

  // -------- pack all 13 weight matrices (transposed f16, zero padded) --------
  const float* mats[13];
  mats[0] = nW2; mats[1] = eW2;
  for (int l = 0; l < LVL; ++l) { mats[2 + l] = mW1 + (size_t)l * DD * DD; }
  for (int l = 0; l < LVL; ++l) { mats[7 + l] = mW2 + (size_t)l * DD * DD; }
  mats[12] = pW;
  for (int i = 0; i < 13; ++i)
    pack_wt_kernel<<<cdiv(DP * DP, 256), 256, 0, stream>>>(mats[i], wt + (size_t)i * WSZ);

  // -------- edge encoder: ef = relu(edge_attr@eW1+eb1) @ eW2 + eb2 --------
  edge_hidden_kernel<<<cdiv(E * DP, 256), 256, 0, stream>>>(eattr, eW1, eb1, bf1, E);
  launch_gemm(bf1, wt + 1 * WSZ, eb2, ef, nullptr, E, 0);

  // -------- node encoder: h = atom_emb[z] + relu([c,f]@nW1+nb1) @ nW2 + nb2 --------
  node_hidden_kernel<<<cdiv(N * DP, 256), 256, 0, stream>>>(chir, fch, nW1, nb1, bf1, N);
  launch_gemm(bf1, wt + 0 * WSZ, nb2, h, nullptr, N, 0);
  add_emb_kernel<<<cdiv(N * DD, 256), 256, 0, stream>>>(h, z, aemb, N);

  // -------- 5 GINE layers --------
  const int CH = DD / 4;
  for (int l = 0; l < LVL; ++l) {
    // xin = h  (the (1+eps)*x_i term, eps=0) then scatter-add messages
    copy_f32_kernel<<<cdiv(N * DD, 256), 256, 0, stream>>>(h, xin, N * DD);
    scatter_msg_kernel<<<cdiv(E * CH, 256), 256, 0, stream>>>(h, ef, src, dst, xin, E);
    // MLP: t = relu(xin @ mW1 + mb1);  h2 = t @ mW2 + mb2   (h2 overwrites xin)
    to_f16_kernel<<<cdiv(N * DP, 256), 256, 0, stream>>>(xin, bf1, N);
    launch_gemm(bf1, wt + (size_t)(2 + l) * WSZ, mb1 + (size_t)l * DD, nullptr, bf2, N, 1);
    launch_gemm(bf2, wt + (size_t)(7 + l) * WSZ, mb2 + (size_t)l * DD, xin, nullptr, N, 0);
    // BatchNorm (training stats) + ReLU -> h
    zero_f32_kernel<<<cdiv(2 * DD, 256), 256, 0, stream>>>(st, 2 * DD);
    bn_stats_kernel<<<cdiv(N, 128), 256, 0, stream>>>(xin, st, N);
    bn_apply_kernel<<<cdiv(N * DD, 256), 256, 0, stream>>>(
        xin, st, gamma + (size_t)l * DD, beta + (size_t)l * DD, h, N, 1.f / (float)N);
  }

  // -------- mean pool + head --------
  zero_f32_kernel<<<cdiv(G * DD, 256), 256, 0, stream>>>(gsum, G * DD);
  zero_f32_kernel<<<cdiv(G, 256), 256, 0, stream>>>(gcnt, G);
  pool_count_kernel<<<cdiv(N, 256), 256, 0, stream>>>(batch, gcnt, N);
  pool_sum_kernel<<<cdiv(N * CH, 256), 256, 0, stream>>>(h, batch, gsum, N);
  pool_div_kernel<<<cdiv(G * DP, 256), 256, 0, stream>>>(gsum, gcnt, pbuf, G);
  launch_gemm(pbuf, wt + 12 * WSZ, pb, out, nullptr, G, 0);
}
